// TransformerBlock_18098992185553
// MI455X (gfx1250) — compile-verified, matrix operand order checked by reading, not run on hardware
//
#include <hip/hip_runtime.h>

// ---------------------------------------------------------------------------
// Problem constants (reference: B=16, S=1024, D=1024, H=16, DK=64)
// ---------------------------------------------------------------------------
constexpr int Bb  = 16;
constexpr int Ss  = 1024;
constexpr int Dd  = 1024;
constexpr int Hh  = 16;
constexpr int DKk = 64;
constexpr int GM  = Bb * Ss;   // 16384 rows of activations
constexpr int GN  = Dd;        // 1024
constexpr int GK  = Dd;        // 1024

constexpr float NEGV  = -4294967295.0f;          // torch masked_fill value
constexpr float SCALE = 1.0f / (8.0f + 1e-6f);   // 1/(sqrt(DK)+eps)

// ---------------------------------------------------------------------------
// Trivial vector types (HIP's float4/uint4 have ctors -> unusable in unions)
// ---------------------------------------------------------------------------
typedef __attribute__((ext_vector_type(16))) __bf16     v16bf;
typedef __attribute__((ext_vector_type(8)))  float      v8f;
typedef __attribute__((ext_vector_type(4)))  unsigned   u32x4;
typedef __attribute__((ext_vector_type(4)))  int        i32x4;
typedef __attribute__((ext_vector_type(4)))  float      f32x4;

union FragU { v16bf v; u32x4 q[2]; };

__device__ inline v8f v8f_zero() {
  v8f z = {0.f, 0.f, 0.f, 0.f, 0.f, 0.f, 0.f, 0.f};
  return z;
}

__device__ inline v8f wmma_bf16(v16bf a, v16bf b, v8f c) {
  // 8 args: (neg_a, A, neg_b, B, c_mod, C, reuse_a, reuse_b)
  return __builtin_amdgcn_wmma_f32_16x16x32_bf16(false, a, false, b,
                                                 (short)0, c, false, false);
}

// ---------------------------------------------------------------------------
// Async global->LDS 16B copy (GLOBAL_LOAD_ASYNC_TO_LDS_B128, ASYNCcnt).
// Builtin param type (from hipcc diagnostic): pointer to 4xi32 vector.
// Tier 2: clang builtin; tier 1: inline asm on gfx1250; tier 0: sync copy.
// ---------------------------------------------------------------------------
#if defined(__gfx1250__)
  #if __has_builtin(__builtin_amdgcn_global_load_async_to_lds_b128)
    #define ASYNC_MODE 2
  #else
    #define ASYNC_MODE 1
  #endif
#else
  #define ASYNC_MODE 0
#endif

__device__ inline void async_copy_b128(const void* g, void* l) {
#if ASYNC_MODE == 2
  typedef __attribute__((address_space(1))) i32x4* gas_p;  // global v4i*
  typedef __attribute__((address_space(3))) i32x4* las_p;  // LDS v4i*
  __builtin_amdgcn_global_load_async_to_lds_b128((gas_p)g, (las_p)l, 0, 0);
#elif ASYNC_MODE == 1
  const unsigned lofs = (unsigned)(unsigned long long)l;  // low 32b = LDS addr
  asm volatile("global_load_async_to_lds_b128 %0, %1, off"
               :: "v"(lofs), "v"(g) : "memory");
#else
  *(u32x4*)l = *(const u32x4*)g;
#endif
}

__device__ inline void async_wait0() {
#if ASYNC_MODE == 2
  #if __has_builtin(__builtin_amdgcn_s_wait_asynccnt)
  __builtin_amdgcn_s_wait_asynccnt(0);
  #else
  asm volatile("s_wait_asynccnt 0x0" ::: "memory");
  #endif
#elif ASYNC_MODE == 1
  asm volatile("s_wait_asynccnt 0x0" ::: "memory");
#endif
}

// Load a 16x32 bf16 operand fragment from an LDS tile laid out row-major
// [row][k] with leading dimension `ld` (halfs, multiple of 8 => 16B rows).
// A-matrix VGPR layout (ISA 7.12.2): lane L holds row (L&15);
// elements 0..7  = K[k0 + 8*(L>=16) .. +7], elements 8..15 = K[+16 .. ].
// B fragments use the same pattern with the tile stored as [n][k].
__device__ inline v16bf ld_frag(const __bf16* base, int ld, int row0, int k0) {
  const int lane = threadIdx.x & 31;
  const int r  = row0 + (lane & 15);
  const int kb = k0 + ((lane >> 4) << 3);
  FragU f;
  f.q[0] = *(const u32x4*)(base + r * ld + kb);
  f.q[1] = *(const u32x4*)(base + r * ld + kb + 16);
  return f.v;
}

// ---------------------------------------------------------------------------
// Generic bf16 WMMA GEMM: C[M=16384][N=1024] = A[M][K=1024] * B
//   TAF   : 0 = A is fp32 (converted to bf16 on load), 1 = A is bf16 (async)
//   TRANSB: 0 = B is [K][N] fp32, 1 = B is [N][K] fp32 (i.e. X @ W.T)
//   EPI   : 0 = none, 1 = +bias, 2 = +bias & ReLU
//   OUTM  : 0 = fp32 row-major, 1 = bf16 row-major,
//           2 = bf16 scatter to [B][H][S][DK] (per-head layout for attention)
// Tile: 128x128 per workgroup, BK=32, double-buffered LDS; 8 waves in 2x4
// grid => 8 v_wmma_f32_16x16x32_bf16 per wave per K-step, staging of tile
// k+1 (async where possible) overlapped with WMMA on tile k.
// ---------------------------------------------------------------------------
template <int TAF, int TRANSB, int EPI, int OUTM>
__global__ __launch_bounds__(256)
void gemm_kernel(const void* __restrict__ Ap, const float* __restrict__ Bw,
                 const float* __restrict__ bias, void* __restrict__ Cp) {
  constexpr int BM = 128, BN = 128, BK = 32, LDT = 40; // 40 halfs = 80B rows
  __shared__ __align__(16) __bf16 Asb[2][BM * LDT];
  __shared__ __align__(16) __bf16 Bsb[2][BN * LDT];

  const int tid  = threadIdx.x;
  const int lane = tid & 31;
  const int wave = tid >> 5;
  const int wr   = wave >> 2;  // 0..1  (row group of 64)
  const int wc   = wave & 3;   // 0..3  (col group of 32)
  const long m0  = (long)blockIdx.y * BM;
  const long n0  = (long)blockIdx.x * BN;

  auto stageA = [&](int buf, int kk) {
    if (TAF == 0) {
      const float* A = (const float*)Ap;
#pragma unroll
      for (int i = 0; i < 4; ++i) {
        const int idx = tid + 256 * i;           // 1024 float4 chunks
        const int row = idx >> 3, c4 = (idx & 7) << 2;
        const f32x4 vA = *(const f32x4*)(A + (m0 + row) * GK + kk + c4);
        __bf16* dst = &Asb[buf][row * LDT + c4];
        dst[0] = (__bf16)vA[0]; dst[1] = (__bf16)vA[1];
        dst[2] = (__bf16)vA[2]; dst[3] = (__bf16)vA[3];
      }
    } else {
      const __bf16* A = (const __bf16*)Ap;
      const int row = tid >> 1, c = (tid & 1) << 4; // 2 x 16B per thread
      const __bf16* src = A + (m0 + row) * GK + kk + c;
      __bf16* dst = &Asb[buf][row * LDT + c];
      async_copy_b128(src, dst);
      async_copy_b128(src + 8, dst + 8);
    }
  };
  auto stageB = [&](int buf, int kk) {
    if (TRANSB == 0) {
      // B is [K][N] fp32; transpose while converting into [n][k]
#pragma unroll
      for (int i = 0; i < 4; ++i) {
        const int idx = tid + 256 * i;           // 32 k * 32 n-groups
        const int k = idx >> 5, n4 = (idx & 31) << 2;
        const f32x4 w = *(const f32x4*)(Bw + (long)(kk + k) * GN + n0 + n4);
        Bsb[buf][(n4 + 0) * LDT + k] = (__bf16)w[0];
        Bsb[buf][(n4 + 1) * LDT + k] = (__bf16)w[1];
        Bsb[buf][(n4 + 2) * LDT + k] = (__bf16)w[2];
        Bsb[buf][(n4 + 3) * LDT + k] = (__bf16)w[3];
      }
    } else {
      // B is [N][K] fp32; direct row copy with conversion
#pragma unroll
      for (int i = 0; i < 4; ++i) {
        const int idx = tid + 256 * i;
        const int n = idx >> 3, c4 = (idx & 7) << 2;
        const f32x4 w = *(const f32x4*)(Bw + (n0 + n) * (long)GK + kk + c4);
        __bf16* dst = &Bsb[buf][n * LDT + c4];
        dst[0] = (__bf16)w[0]; dst[1] = (__bf16)w[1];
        dst[2] = (__bf16)w[2]; dst[3] = (__bf16)w[3];
      }
    }
  };

  v8f acc[4][2];
#pragma unroll
  for (int i = 0; i < 4; ++i) { acc[i][0] = v8f_zero(); acc[i][1] = v8f_zero(); }

  constexpr int NK = GK / BK;   // 32 K-steps
  stageA(0, 0);
  stageB(0, 0);
  async_wait0();
  __syncthreads();

  for (int kt = 0; kt < NK; ++kt) {
    const int cur = kt & 1;
    if (kt + 1 < NK) {          // stage next tile while computing this one
      stageA(cur ^ 1, (kt + 1) * BK);
      stageB(cur ^ 1, (kt + 1) * BK);
    }
    const v16bf bf0 = ld_frag(&Bsb[cur][0], LDT, wc * 32, 0);
    const v16bf bf1 = ld_frag(&Bsb[cur][0], LDT, wc * 32 + 16, 0);
#pragma unroll
    for (int mt = 0; mt < 4; ++mt) {
      const v16bf af = ld_frag(&Asb[cur][0], LDT, wr * 64 + mt * 16, 0);
      acc[mt][0] = wmma_bf16(af, bf0, acc[mt][0]);
      acc[mt][1] = wmma_bf16(af, bf1, acc[mt][1]);
    }
    async_wait0();
    __syncthreads();
  }

  // ---- epilogue: C layout (ISA): lane holds n=(lane&15), VGPR r holds
  // m = r + 8*(lane>=16) ----
  const int hi = lane >> 4, ln = lane & 15;
#pragma unroll
  for (int mt = 0; mt < 4; ++mt)
#pragma unroll
    for (int nt = 0; nt < 2; ++nt)
#pragma unroll
      for (int r = 0; r < 8; ++r) {
        const long m = m0 + wr * 64 + mt * 16 + r + hi * 8;
        const long n = n0 + wc * 32 + nt * 16 + ln;
        float v = acc[mt][nt][r];
        if (EPI >= 1) v += bias[n];
        if (EPI == 2) v = fmaxf(v, 0.f);
        if (OUTM == 0) {
          ((float*)Cp)[m * GN + n] = v;
        } else if (OUTM == 1) {
          ((__bf16*)Cp)[m * GN + n] = (__bf16)v;
        } else {
          const long b = m / Ss, s = m % Ss;
          const long h = n / DKk, dk = n % DKk;
          ((__bf16*)Cp)[(((b * Hh + h) * Ss) + s) * DKk + dk] = (__bf16)v;
        }
      }
}

// ---------------------------------------------------------------------------
// Flash attention: one workgroup per (b,h, 128-row q block); 8 waves, each
// owns 16 q rows. K/V streamed in 64-row blocks (causal upper bound), online
// softmax in fp32. Scores and P@V use v_wmma_f32_16x16x32_bf16; Q/K tiles
// staged with async global->LDS; P converts C-layout -> A-layout through a
// per-wave LDS staging tile.
// ---------------------------------------------------------------------------
__global__ __launch_bounds__(256)
void attn_kernel(const __bf16* __restrict__ Qh, const __bf16* __restrict__ Kh,
                 const __bf16* __restrict__ Vh,
                 const unsigned char* __restrict__ maskp,
                 __bf16* __restrict__ ctx) {
  constexpr int QT = 128, KT = 64, LQ = 72, LK = 72, LV = 72, LP = 72;
  __shared__ __align__(16) __bf16 Qs[QT * LQ];      // [qrow][dk]    18.0 KB
  __shared__ __align__(16) __bf16 Ks[KT * LK];      // [krow][dk]     9.0 KB
  __shared__ __align__(16) __bf16 Vt[DKk * LV];     // [dk][krow]     9.0 KB
  __shared__ __align__(16) __bf16 Ps[8 * 16 * LP];  // per-wave P    18.0 KB

  const int tid  = threadIdx.x;
  const int lane = tid & 31;
  const int wave = tid >> 5;
  const int hi   = lane >> 4, ln = lane & 15;
  const int qb   = blockIdx.x;           // 0..S/128-1
  const int bh   = blockIdx.y;           // b*H + h
  const int b    = bh >> 4, h = bh & 15;
  const int qrow_base = qb * QT + wave * 16;

  // ---- stage Q tile (async DMA into LDS) ----
  {
    const int row = tid >> 1, c = (tid & 1) << 5;   // 32 halfs / thread
    const __bf16* src = Qh + ((long)bh * Ss + qb * QT + row) * DKk + c;
    __bf16* dst = &Qs[row * LQ + c];
#pragma unroll
    for (int i = 0; i < 4; ++i) async_copy_b128(src + 8 * i, dst + 8 * i);
  }
  async_wait0();
  __syncthreads();

  const v16bf aq0 = ld_frag(Qs, LQ, wave * 16, 0);
  const v16bf aq1 = ld_frag(Qs, LQ, wave * 16, 32);

  float mrow[8], lrow[8];
  bool  mq[8];
  v8f   o[4];
#pragma unroll
  for (int r = 0; r < 8; ++r) {
    const int q = qrow_base + r + hi * 8;
    mq[r]   = maskp[(long)b * Ss + q] != 0;   // padded query row
    mrow[r] = -INFINITY;
    lrow[r] = 0.f;
  }
#pragma unroll
  for (int nt = 0; nt < 4; ++nt) o[nt] = v8f_zero();

  const int jmax = 2 * qb + 1;                // causal block bound
  for (int j = 0; j <= jmax; ++j) {
    __syncthreads();
    // ---- stage K block [64][64] via async DMA ----
    {
      const int row = tid >> 2, c = (tid & 3) << 4;
      const __bf16* src = Kh + ((long)bh * Ss + j * KT + row) * DKk + c;
      __bf16* dst = &Ks[row * LK + c];
      async_copy_b128(src, dst);
      async_copy_b128(src + 8, dst + 8);
    }
    // ---- stage V block transposed into [dk][krow] (converting scatter) ----
#pragma unroll
    for (int i = 0; i < 2; ++i) {
      const int idx = tid + 256 * i;          // 64 s * 8 dk-groups
      const int s = idx >> 3, dg = (idx & 7) << 3;
      union { u32x4 q; __bf16 e[8]; } u;
      u.q = *(const u32x4*)(Vh + ((long)bh * Ss + j * KT + s) * DKk + dg);
#pragma unroll
      for (int e = 0; e < 8; ++e) Vt[(dg + e) * LV + s] = u.e[e];
    }
    async_wait0();
    __syncthreads();

    // ---- scores: S = Q(16x64) @ K^T(64x64) ----
    v8f st[4];
#pragma unroll
    for (int nt = 0; nt < 4; ++nt) {
      v8f c = v8f_zero();
      c = wmma_bf16(aq0, ld_frag(Ks, LK, nt * 16, 0), c);
      c = wmma_bf16(aq1, ld_frag(Ks, LK, nt * 16, 32), c);
      st[nt] = c;
    }

    // ---- online softmax per row ----
    __bf16* pw = &Ps[wave * 16 * LP];
#pragma unroll
    for (int r = 0; r < 8; ++r) {
      const int q = qrow_base + r + hi * 8;
#pragma unroll
      for (int nt = 0; nt < 4; ++nt) {
        const int k = j * KT + nt * 16 + ln;
        float v = st[nt][r] * SCALE;
        if (k > q || mq[r]) v = NEGV;         // causal | pad (NEG as reference)
        st[nt][r] = v;
      }
      float t = fmaxf(fmaxf(st[0][r], st[1][r]), fmaxf(st[2][r], st[3][r]));
#pragma unroll
      for (int off = 1; off < 16; off <<= 1) t = fmaxf(t, __shfl_xor(t, off));
      const float nm    = fmaxf(mrow[r], t);
      const float alpha = __expf(mrow[r] - nm);
      float rs = 0.f;
#pragma unroll
      for (int nt = 0; nt < 4; ++nt) {
        const float p = __expf(st[nt][r] - nm);
        st[nt][r] = p;
        rs += p;
      }
#pragma unroll
      for (int off = 1; off < 16; off <<= 1) rs += __shfl_xor(rs, off);
      lrow[r] = lrow[r] * alpha + rs;
      mrow[r] = nm;
#pragma unroll
      for (int nt = 0; nt < 4; ++nt) o[nt][r] = o[nt][r] * alpha;
      // stage P row into per-wave LDS tile ([row][k] => A-layout readable)
#pragma unroll
      for (int nt = 0; nt < 4; ++nt)
        pw[(r + hi * 8) * LP + nt * 16 + ln] = (__bf16)st[nt][r];
    }
    asm volatile("s_wait_dscnt 0" ::: "memory");  // P staging visible in-wave

    // ---- O += P(16x64) @ V(64x64) ----
#pragma unroll
    for (int ks = 0; ks < 2; ++ks) {
      const v16bf pa = ld_frag(pw, LP, 0, ks * 32);
#pragma unroll
      for (int nt = 0; nt < 4; ++nt)
        o[nt] = wmma_bf16(pa, ld_frag(Vt, LV, nt * 16, ks * 32), o[nt]);
    }
  }

  // ---- normalize and scatter ctx back to [B][S][D] (merging heads) ----
#pragma unroll
  for (int nt = 0; nt < 4; ++nt)
#pragma unroll
    for (int r = 0; r < 8; ++r) {
      const int q = qrow_base + r + hi * 8;
      const float v = o[nt][r] / lrow[r];
      ctx[((long)b * Ss + q) * Dd + h * DKk + nt * 16 + ln] = (__bf16)v;
    }
}

// ---------------------------------------------------------------------------
// Fused residual + LayerNorm; optional bf16 mirror for the next bf16 GEMM.
// One workgroup (256 threads) per 1024-wide row.
// ---------------------------------------------------------------------------
__global__ __launch_bounds__(256)
void add_ln_kernel(const float* __restrict__ Rr, const float* __restrict__ Yy,
                   const float* __restrict__ w, const float* __restrict__ bv,
                   float* __restrict__ outF, __bf16* __restrict__ outB) {
  __shared__ float red[8];
  const long row = blockIdx.x;
  const int  tid = threadIdx.x;

  float v[4];
  float s = 0.f;
#pragma unroll
  for (int i = 0; i < 4; ++i) {
    const int c = tid + (i << 8);
    const float t = Rr[row * Dd + c] + Yy[row * Dd + c];
    v[i] = t;
    s += t;
  }
#pragma unroll
  for (int off = 1; off < 32; off <<= 1) s += __shfl_xor(s, off);
  if ((tid & 31) == 0) red[tid >> 5] = s;
  __syncthreads();
  float tot = 0.f;
#pragma unroll
  for (int i = 0; i < 8; ++i) tot += red[i];
  const float mu = tot * (1.0f / Dd);
  __syncthreads();

  float s2 = 0.f;
#pragma unroll
  for (int i = 0; i < 4; ++i) { const float d = v[i] - mu; s2 += d * d; }
#pragma unroll
  for (int off = 1; off < 32; off <<= 1) s2 += __shfl_xor(s2, off);
  if ((tid & 31) == 0) red[tid >> 5] = s2;
  __syncthreads();
  float tot2 = 0.f;
#pragma unroll
  for (int i = 0; i < 8; ++i) tot2 += red[i];
  const float rstd = rsqrtf(tot2 * (1.0f / Dd) + 1e-5f);

#pragma unroll
  for (int i = 0; i < 4; ++i) {
    const int c = tid + (i << 8);
    const float ov = (v[i] - mu) * rstd * w[c] + bv[c];
    outF[row * Dd + c] = ov;
    if (outB) outB[row * Dd + c] = (__bf16)ov;
  }
}

// ---------------------------------------------------------------------------
// Host-side orchestration. Workspace layout (bytes), with aggressive reuse:
//   [0,32M)    Qh bf16 [B,H,S,DK]     -> later: X fp32 low half
//   [32M,64M)  Kh bf16                -> later: X fp32 high half
//   [64M,96M)  Vh bf16                -> later: X bf16
//   [96M,128M) ctx bf16 [B,S,D]       -> later: H1 bf16
//   [128M,192M) Y fp32 [B,S,D]        -> later: Y2 fp32
// Peak = 192 MB.
// ---------------------------------------------------------------------------
extern "C" void kernel_launch(void* const* d_in, const int* in_sizes, int n_in,
                              void* d_out, int out_size, void* d_ws,
                              size_t ws_size, hipStream_t stream) {
  (void)in_sizes; (void)n_in; (void)out_size; (void)ws_size;

  const float* Qin  = (const float*)d_in[0];
  const float* Kin  = (const float*)d_in[1];
  const float* Vin  = (const float*)d_in[2];
  const unsigned char* mask = (const unsigned char*)d_in[3];
  const float* W_q  = (const float*)d_in[4];
  const float* W_k  = (const float*)d_in[5];
  const float* W_v  = (const float*)d_in[6];
  const float* W_o  = (const float*)d_in[7];
  const float* l1w  = (const float*)d_in[8];
  const float* l1b  = (const float*)d_in[9];
  const float* l2w  = (const float*)d_in[10];
  const float* l2b  = (const float*)d_in[11];
  const float* lnw  = (const float*)d_in[12];
  const float* lnb  = (const float*)d_in[13];

  char* ws = (char*)d_ws;
  const size_t SZ = (size_t)GM * GN * sizeof(__bf16);  // 32 MB
  __bf16* qh  = (__bf16*)(ws + 0 * SZ);
  __bf16* kh  = (__bf16*)(ws + 1 * SZ);
  __bf16* vh  = (__bf16*)(ws + 2 * SZ);
  __bf16* ctx = (__bf16*)(ws + 3 * SZ);
  float*  Yb  = (float*)(ws + 4 * SZ);   // 64 MB fp32
  float*  Xf  = (float*)(ws + 0 * SZ);   // reuse qh+kh (dead after attention)
  __bf16* Xb  = (__bf16*)(ws + 2 * SZ);  // reuse vh
  __bf16* H1  = (__bf16*)(ws + 3 * SZ);  // reuse ctx (dead after out-proj)
  float*  Y2  = (float*)(ws + 4 * SZ);   // reuse Yb (dead after LN1)

  const dim3 gg(GN / 128, GM / 128);     // (8, 128)
  const dim3 blk(256);

  // QKV projections: fp32 in, bf16 per-head [B,H,S,DK] out
  gemm_kernel<0, 0, 0, 2><<<gg, blk, 0, stream>>>(Qin, W_q, nullptr, qh);
  gemm_kernel<0, 0, 0, 2><<<gg, blk, 0, stream>>>(Kin, W_k, nullptr, kh);
  gemm_kernel<0, 0, 0, 2><<<gg, blk, 0, stream>>>(Vin, W_v, nullptr, vh);

  // Flash attention -> ctx bf16 [B,S,D]
  attn_kernel<<<dim3(Ss / 128, Bb * Hh), blk, 0, stream>>>(qh, kh, vh, mask,
                                                           ctx);

  // Output projection -> Y fp32
  gemm_kernel<1, 0, 0, 0><<<gg, blk, 0, stream>>>(ctx, W_o, nullptr, Yb);

  // X = LN(Q + Y), keep fp32 + bf16 copies
  add_ln_kernel<<<GM, blk, 0, stream>>>(Qin, Yb, lnw, lnb, Xf, Xb);

  // FFN: H1 = relu(X @ W1^T + b1) -> bf16 ; Y2 = H1 @ W2^T + b2 -> fp32
  gemm_kernel<1, 1, 2, 1><<<gg, blk, 0, stream>>>(Xb, l1w, l1b, H1);
  gemm_kernel<1, 1, 1, 0><<<gg, blk, 0, stream>>>(H1, l2w, l2b, Y2);

  // out = LN(Y2 + X)
  add_ln_kernel<<<GM, blk, 0, stream>>>(Xf, Y2, lnw, lnb, (float*)d_out,
                                        nullptr);
}
